// RSGNN_10694468567404
// MI455X (gfx1250) — matrix operand order repaired and checked by reading, not compiled, not run on hardware
//
#include <hip/hip_runtime.h>
#include <hip/hip_bf16.h>

#define NN 65536
#define FF 512
#define HH 256
#define EE 1048576
#define RR 512

typedef __attribute__((ext_vector_type(16))) _Float16 v16h;
typedef __attribute__((ext_vector_type(8)))  _Float16 v8h;
typedef __attribute__((ext_vector_type(8)))  float    v8f;
typedef __attribute__((ext_vector_type(4)))  float    v4f;

// ---------------- utility kernels ----------------

__global__ void k_zero_f32(float* p, long n) {
    long i = (long)blockIdx.x * blockDim.x + threadIdx.x;
    long stride = (long)gridDim.x * blockDim.x;
    for (; i < n; i += stride) p[i] = 0.0f;
}

__global__ void k_init_small(unsigned long long* packed, float* loss_slot) {
    int i = blockIdx.x * blockDim.x + threadIdx.x;
    if (i < RR) packed[i] = 0xFFFFFFFFFFFFFFFFULL;
    if (i == 0) *loss_slot = 0.0f;
}

// Pack a K x Ncols f32 matrix into WMMA B-fragment order (f16).
// element (k,n) = transpose ? src[n*srcStride + k] : src[k*srcStride + n]
// B fragment layout per 32x16 tile: lane = ((k%32)/8%2)*16 + n%16, slot i = k%8 + (k%32)/16*8.
__global__ void k_pack_b(const float* __restrict__ src, _Float16* __restrict__ dst,
                         int K, int Ncols, int srcStride, int transpose) {
    int idx = blockIdx.x * blockDim.x + threadIdx.x;
    if (idx >= K * Ncols) return;
    int k = idx / Ncols, n = idx % Ncols;
    float val = transpose ? src[(size_t)n * srcStride + k] : src[(size_t)k * srcStride + n];
    int kt = k >> 5, kk = k & 31, half = (kk >> 3) & 1, i = (kk & 7) + ((kk >> 4) << 3);
    int nt = n >> 4, nn = n & 15;
    int NT = Ncols >> 4;
    int lane = half * 16 + nn;
    dst[((((size_t)kt * NT + nt) * 32 + lane) << 4) + i] = (_Float16)val;
}

__global__ void k_degree(const int* __restrict__ snd, const int* __restrict__ rcv,
                         float* __restrict__ ds, float* __restrict__ dr) {
    int e = blockIdx.x * blockDim.x + threadIdx.x;
    if (e >= EE) return;
    atomicAdd(&ds[snd[e]], 1.0f);
    atomicAdd(&dr[rcv[e]], 1.0f);
}

__global__ void k_rsqrt_inplace(float* d, int n) {
    int i = blockIdx.x * blockDim.x + threadIdx.x;
    if (i < n) d[i] = rsqrtf(fmaxf(d[i], 1.0f));
}

// ---------------- GEMM: Y[N,H] = X[N,F] @ W (packed f16 frags) + bias ----------------
// block = 256 threads = 8 waves; wave computes 16 rows x 64 cols.
// grid = (N/128, H/64)
__global__ void k_gemm_xw(const float* __restrict__ X, const _Float16* __restrict__ Wp,
                          const float* __restrict__ bias, float* __restrict__ Y) {
    const int lane = threadIdx.x & 31;
    const int wv   = threadIdx.x >> 5;
    const int half = lane >> 4;
    const int row  = blockIdx.x * 128 + wv * 16 + (lane & 15);
    const int NT   = HH / 16;

    v8f acc0 = {}, acc1 = {}, acc2 = {}, acc3 = {};

    for (int kt = 0; kt < FF / 32; ++kt) {
        // A fragment: row = lane%16, K chunks at kt*32 + half*8 (+0..7) and +16
        const float* xp = X + (size_t)row * FF + kt * 32 + half * 8;
        v4f a0 = *(const v4f*)(xp);
        v4f a1 = *(const v4f*)(xp + 4);
        v4f a2 = *(const v4f*)(xp + 16);
        v4f a3 = *(const v4f*)(xp + 20);
        v16h afr;
#pragma unroll
        for (int i = 0; i < 4; ++i) {
            afr[i]      = (_Float16)a0[i];
            afr[4 + i]  = (_Float16)a1[i];
            afr[8 + i]  = (_Float16)a2[i];
            afr[12 + i] = (_Float16)a3[i];
        }
        const int nbase = blockIdx.y * 4;
        const v16h b0 = *(const v16h*)(Wp + ((((size_t)kt * NT + nbase + 0) * 32 + lane) << 4));
        const v16h b1 = *(const v16h*)(Wp + ((((size_t)kt * NT + nbase + 1) * 32 + lane) << 4));
        const v16h b2 = *(const v16h*)(Wp + ((((size_t)kt * NT + nbase + 2) * 32 + lane) << 4));
        const v16h b3 = *(const v16h*)(Wp + ((((size_t)kt * NT + nbase + 3) * 32 + lane) << 4));
        acc0 = __builtin_amdgcn_wmma_f32_16x16x32_f16(false, afr, false, b0, (short)0, acc0, false, false);
        acc1 = __builtin_amdgcn_wmma_f32_16x16x32_f16(false, afr, false, b1, (short)0, acc1, false, false);
        acc2 = __builtin_amdgcn_wmma_f32_16x16x32_f16(false, afr, false, b2, (short)0, acc2, false, false);
        acc3 = __builtin_amdgcn_wmma_f32_16x16x32_f16(false, afr, false, b3, (short)0, acc3, false, false);
    }

    // C layout: VGPR g -> row (half*8+g), lane%16 -> col
    const int rowbase = blockIdx.x * 128 + wv * 16 + half * 8;
    v8f accs[4] = {acc0, acc1, acc2, acc3};
#pragma unroll
    for (int nt = 0; nt < 4; ++nt) {
        int col = blockIdx.y * 64 + nt * 16 + (lane & 15);
        float bb = bias[col];
#pragma unroll
        for (int g = 0; g < 8; ++g)
            Y[(size_t)(rowbase + g) * HH + col] = accs[nt][g] + bb;
    }
}

// ---------------- scatter: agg[r] += xw[s] * rsd_s[s], one block per edge ----------------
__global__ void k_scatter(const float* __restrict__ xw, const float* __restrict__ rsd_s,
                          const int* __restrict__ snd, const int* __restrict__ rcv,
                          float* __restrict__ agg) {
    int e = blockIdx.x;
    int t = threadIdx.x;
    int s = snd[e], r = rcv[e];
    float sc = rsd_s[s];
    atomicAdd(&agg[(size_t)r * HH + t], xw[(size_t)s * HH + t] * sc);
}

// ---------------- finalize GCN: in-place agg *= rsd_r[row]; selu ----------------
__global__ void k_finalize(float* agg, const float* __restrict__ rsd_r) {
    size_t idx = (size_t)blockIdx.x * blockDim.x + threadIdx.x;
    if (idx >= (size_t)NN * HH) return;
    int row = (int)(idx >> 8);  // H == 256
    float x = agg[idx] * rsd_r[row];
    const float scale = 1.0507009873554805f, alpha = 1.6732632423543772f;
    float y = (x > 0.0f) ? scale * x : scale * alpha * (__expf(x) - 1.0f);
    agg[idx] = y;
}

// ---------------- column sums of nodes1 ----------------
__global__ void k_colsum(const float* __restrict__ n1, float* __restrict__ colsum) {
    int t = threadIdx.x;             // column
    int rb = blockIdx.x * 256;       // row chunk
    float acc = 0.0f;
    for (int rr = 0; rr < 256; ++rr) acc += n1[(size_t)(rb + rr) * HH + t];
    atomicAdd(&colsum[t], acc);
}

// ---------------- summary + v = W_b @ sigmoid(mean) ----------------
__global__ void k_summary_v(const float* __restrict__ colsum, const float* __restrict__ Wb,
                            float* __restrict__ v) {
    __shared__ float s[HH];
    int t = threadIdx.x;
    float m = colsum[t] * (1.0f / (float)NN);
    s[t] = 1.0f / (1.0f + __expf(-m));
    __syncthreads();
    float acc = 0.0f;
    for (int j = 0; j < HH; ++j) acc += Wb[(size_t)t * HH + j] * s[j];
    v[t] = acc;
}

// ---------------- logits = concat(n1,n2) @ v ; one wave (32 lanes) per row ----------------
__global__ void k_logits(const float* __restrict__ n1, const float* __restrict__ n2,
                         const float* __restrict__ v, float* __restrict__ out) {
    int lane = threadIdx.x & 31;
    int wv = threadIdx.x >> 5;
    int row = blockIdx.x * 8 + wv;
    const float* src = (row < NN) ? (n1 + (size_t)row * HH) : (n2 + (size_t)(row - NN) * HH);
    float acc = 0.0f;
    for (int j = lane; j < HH; j += 32) acc += src[j] * v[j];
#pragma unroll
    for (int m = 16; m >= 1; m >>= 1) acc += __shfl_xor(acc, m, 32);
    if (lane == 0) out[row] = acc;
}

// ---------------- emb: row-normalize nodes1 -> f32 out + f16 copy ----------------
__global__ void k_emb(const float* __restrict__ n1, float* __restrict__ emb,
                      _Float16* __restrict__ embh) {
    int lane = threadIdx.x & 31;
    int wv = threadIdx.x >> 5;
    int row = blockIdx.x * 8 + wv;
    const float* src = n1 + (size_t)row * HH;
    float sq = 0.0f;
    for (int j = lane; j < HH; j += 32) { float x = src[j]; sq += x * x; }
#pragma unroll
    for (int m = 16; m >= 1; m >>= 1) sq += __shfl_xor(sq, m, 32);
    float inv = 1.0f / sqrtf(sq);
    for (int j = lane; j < HH; j += 32) {
        float val = src[j] * inv;
        emb[(size_t)row * HH + j] = val;
        embh[(size_t)row * HH + j] = (_Float16)val;
    }
}

// ---------------- c2[r] = ||centers[r]||^2 ----------------
__global__ void k_c2(const float* __restrict__ centers, float* __restrict__ c2) {
    int lane = threadIdx.x & 31;
    int wv = threadIdx.x >> 5;
    int r = blockIdx.x * 8 + wv;
    const float* src = centers + (size_t)r * HH;
    float sq = 0.0f;
    for (int j = lane; j < HH; j += 32) { float x = src[j]; sq += x * x; }
#pragma unroll
    for (int m = 16; m >= 1; m >>= 1) sq += __shfl_xor(sq, m, 32);
    if (lane == 0) c2[r] = sq;
}

// ---------------- fused clustering GEMM: dists + row-min(sum) + col-argmin ----------------
// block = 256 threads = 8 waves; block covers 16 rows x all 512 centers.
// grid = N/16. dist^2 = 1 + c2 - 2 * emb.centers (||emb||=1 exactly)
__global__ void k_cluster(const _Float16* __restrict__ embh, const _Float16* __restrict__ Cp,
                          const float* __restrict__ c2, unsigned long long* __restrict__ packed,
                          float* __restrict__ loss) {
    const int lane = threadIdx.x & 31;
    const int wv   = threadIdx.x >> 5;
    const int half = lane >> 4;
    const int row16 = blockIdx.x * 16;
    const int arow  = row16 + (lane & 15);
    const int NT = RR / 16;  // 32

    v8f acc[4] = {{}, {}, {}, {}};
    for (int kt = 0; kt < HH / 32; ++kt) {
        const _Float16* ap = embh + (size_t)arow * HH + kt * 32 + half * 8;
        v8h a0 = *(const v8h*)(ap);
        v8h a1 = *(const v8h*)(ap + 16);
        v16h afr;
#pragma unroll
        for (int i = 0; i < 8; ++i) { afr[i] = a0[i]; afr[8 + i] = a1[i]; }
#pragma unroll
        for (int nt = 0; nt < 4; ++nt) {
            int ntile = wv * 4 + nt;
            const v16h bfr = *(const v16h*)(Cp + ((((size_t)kt * NT + ntile) * 32 + lane) << 4));
            acc[nt] = __builtin_amdgcn_wmma_f32_16x16x32_f16(false, afr, false, bfr, (short)0, acc[nt], false, false);
        }
    }

    __shared__ float rmin_lds[8 * 16];
    float rmin[8];
#pragma unroll
    for (int g = 0; g < 8; ++g) rmin[g] = 1e30f;

#pragma unroll
    for (int nt = 0; nt < 4; ++nt) {
        int col = wv * 64 + nt * 16 + (lane & 15);
        float cc = c2[col];
        float best = 1e30f; int bestg = 0;
#pragma unroll
        for (int g = 0; g < 8; ++g) {
            float dist = sqrtf(fmaxf(1.0f + cc - 2.0f * acc[nt][g], 1e-12f));
            rmin[g] = fminf(rmin[g], dist);
            if (dist < best) { best = dist; bestg = g; }
        }
        // combine the two half-wave rows for this column (same col in lane, lane^16)
        int bestrow = row16 + half * 8 + bestg;
        float obest = __shfl_xor(best, 16, 32);
        int orow = __shfl_xor(bestrow, 16, 32);
        if (obest < best || (obest == best && orow < bestrow)) { best = obest; bestrow = orow; }
        unsigned long long pk =
            ((unsigned long long)__float_as_uint(best) << 32) | (unsigned int)bestrow;
        atomicMin(&packed[col], pk);
    }

    // per-row min within half-wave (16 cols), xor 8/4/2/1 stays inside the half
#pragma unroll
    for (int m = 8; m >= 1; m >>= 1)
#pragma unroll
        for (int g = 0; g < 8; ++g) rmin[g] = fminf(rmin[g], __shfl_xor(rmin[g], m, 32));

    if ((lane & 15) == 0) {
#pragma unroll
        for (int g = 0; g < 8; ++g) rmin_lds[wv * 16 + half * 8 + g] = rmin[g];
    }
    __syncthreads();
    if (threadIdx.x < 16) {
        float m = rmin_lds[threadIdx.x];
#pragma unroll
        for (int w = 1; w < 8; ++w) m = fminf(m, rmin_lds[w * 16 + threadIdx.x]);
        atomicAdd(loss, m);  // cluster_loss = sum of per-node min dists
    }
}

__global__ void k_rep(const unsigned long long* __restrict__ packed, float* __restrict__ rep) {
    int r = blockIdx.x * blockDim.x + threadIdx.x;
    if (r < RR) rep[r] = (float)(unsigned int)(packed[r] & 0xFFFFFFFFULL);
}

// ---------------- launch ----------------
extern "C" void kernel_launch(void* const* d_in, const int* in_sizes, int n_in,
                              void* d_out, int out_size, void* d_ws, size_t ws_size,
                              hipStream_t stream) {
    const float* nodes    = (const float*)d_in[0];
    const float* c_nodes  = (const float*)d_in[1];
    const int*   senders  = (const int*)d_in[2];
    const int*   receivers= (const int*)d_in[3];
    const float* W        = (const float*)d_in[4];
    const float* b        = (const float*)d_in[5];
    const float* Wb       = (const float*)d_in[6];
    const float* centers  = (const float*)d_in[7];

    float* out = (float*)d_out;
    // output layout: emb[N*H], centers[R*H], rep_ids[R], cluster_loss[1], logits[2N]
    float* out_emb    = out;
    float* out_cent   = out + (size_t)NN * HH;
    float* out_rep    = out_cent + (size_t)RR * HH;
    float* out_loss   = out_rep + RR;
    float* out_logits = out_loss + 1;

    // workspace layout
    char* w = (char*)d_ws;
    unsigned long long* packed = (unsigned long long*)w;           // R * 8
    float* deg_s  = (float*)(w + RR * 8);                          // N
    float* deg_r  = deg_s + NN;                                    // N
    float* colsum = deg_r + NN;                                    // H
    float* vvec   = colsum + HH;                                   // H
    float* c2     = vvec + HH;                                     // R
    _Float16* packedW = (_Float16*)(c2 + RR);                      // F*H halves
    _Float16* packedC = packedW + (size_t)FF * HH;                 // H*R halves
    float* xw1  = (float*)(packedC + (size_t)HH * RR);             // N*H
    float* xw2  = xw1 + (size_t)NN * HH;                           // N*H
    float* agg1 = xw2 + (size_t)NN * HH;                           // N*H (becomes nodes1)
    float* agg2 = agg1 + (size_t)NN * HH;                          // N*H (becomes nodes2)
    _Float16* embh = (_Float16*)xw1;                               // alias: xw1 free after scatter

    // 1) zero deg_s/deg_r/colsum (contiguous) and agg1/agg2 (contiguous)
    k_zero_f32<<<2048, 256, 0, stream>>>(deg_s, (long)(2 * NN + HH));
    k_zero_f32<<<4096, 256, 0, stream>>>(agg1, (long)2 * NN * HH);
    k_init_small<<<(RR + 255) / 256, 256, 0, stream>>>(packed, out_loss);

    // 2) pack W and centers^T into WMMA B-fragment layouts (f16)
    k_pack_b<<<(FF * HH + 255) / 256, 256, 0, stream>>>(W, packedW, FF, HH, HH, 0);
    k_pack_b<<<(HH * RR + 255) / 256, 256, 0, stream>>>(centers, packedC, HH, RR, HH, 1);

    // 3) degrees, then rsqrt(max(deg,1)) in place (deg_s/deg_r contiguous)
    k_degree<<<EE / 256, 256, 0, stream>>>(senders, receivers, deg_s, deg_r);
    k_rsqrt_inplace<<<(2 * NN + 255) / 256, 256, 0, stream>>>(deg_s, 2 * NN);

    // 4) GCN dense: xw = X @ W + b via WMMA f16->f32
    dim3 ggrid(NN / 128, HH / 64);
    k_gemm_xw<<<ggrid, 256, 0, stream>>>(nodes,   packedW, b, xw1);
    k_gemm_xw<<<ggrid, 256, 0, stream>>>(c_nodes, packedW, b, xw2);

    // 5) scatter-sum with sender normalization
    k_scatter<<<EE, HH, 0, stream>>>(xw1, deg_s, senders, receivers, agg1);
    k_scatter<<<EE, HH, 0, stream>>>(xw2, deg_s, senders, receivers, agg2);

    // 6) receiver normalization + SeLU (in place -> nodes1/nodes2)
    k_finalize<<<(NN * HH) / 256, 256, 0, stream>>>(agg1, deg_r);
    k_finalize<<<(NN * HH) / 256, 256, 0, stream>>>(agg2, deg_r);

    // 7) summary = sigmoid(mean(nodes1)); v = W_bilinear @ summary
    k_colsum<<<NN / 256, 256, 0, stream>>>(agg1, colsum);
    k_summary_v<<<1, HH, 0, stream>>>(colsum, Wb, vvec);

    // 8) logits = concat(nodes1, nodes2) @ v
    k_logits<<<(2 * NN) / 8, 256, 0, stream>>>(agg1, agg2, vvec, out_logits);

    // 9) emb = row-normalize(nodes1) -> d_out + f16 copy for the clustering GEMM
    k_emb<<<NN / 8, 256, 0, stream>>>(agg1, out_emb, embh);

    // 10) centers passthrough output + ||c||^2
    hipMemcpyAsync(out_cent, centers, (size_t)RR * HH * sizeof(float),
                   hipMemcpyDeviceToDevice, stream);
    k_c2<<<RR / 8, 256, 0, stream>>>(centers, c2);

    // 11) fused clustering GEMM (dists + min-sum + argmin)
    k_cluster<<<NN / 16, 256, 0, stream>>>(embh, packedC, c2, packed, out_loss);

    // 12) rep_ids
    k_rep<<<(RR + 255) / 256, 256, 0, stream>>>(packed, out_rep);
}
